// MultiHeadAttention_39341900431603
// MI455X (gfx1250) — matrix operand order repaired and checked by reading, not compiled, not run on hardware
//
#include <hip/hip_runtime.h>
#include <hip/hip_bf16.h>

#define D_MODEL 1024
#define SEQ     2048
#define HEADS   16
#define DH      64
#define BATCH   2
#define MROWS   (BATCH * SEQ)   // 4096

typedef __attribute__((ext_vector_type(16))) __bf16 v16bf;
typedef __attribute__((ext_vector_type(8)))  __bf16 bf16x8;
typedef __attribute__((ext_vector_type(8)))  float  v8f;
typedef __attribute__((ext_vector_type(4)))  unsigned int u32x4;
typedef __attribute__((ext_vector_type(8)))  int i32x8;
typedef __attribute__((ext_vector_type(4)))  int i32x4;

#if defined(__has_builtin)
#if __has_builtin(__builtin_amdgcn_tensor_load_to_lds) && __has_builtin(__builtin_amdgcn_s_wait_tensorcnt)
#define HAVE_TDM 1
#endif
#endif
#ifndef HAVE_TDM
#define HAVE_TDM 0
#endif

template <typename T> struct is_f32      { static constexpr bool v = false; };
template <>           struct is_f32<float>{ static constexpr bool v = true;  };

// fp32 -> bf16, round-to-nearest-even
__device__ inline __bf16 f2bf(float f) {
  unsigned u = __builtin_bit_cast(unsigned, f);
  u = (u + 0x7FFFu + ((u >> 16) & 1u)) >> 16;
  unsigned short s = (unsigned short)u;
  return __builtin_bit_cast(__bf16, s);
}

__device__ inline bf16x8 ld8(const __bf16* p) { return *(const bf16x8*)p; }

__device__ inline v16bf cat8(bf16x8 lo, bf16x8 hi) {
  union U { bf16x8 p[2]; v16bf v; } u;
  u.p[0] = lo; u.p[1] = hi;
  return u.v;
}

__device__ inline bf16x8 load8_cvt(const float* src) {
  bf16x8 v;
  for (int i = 0; i < 8; ++i) v[i] = f2bf(src[i]);
  return v;
}
__device__ inline bf16x8 load8_cvt(const __bf16* src) { return *(const bf16x8*)src; }

// Low 32 bits of a generic pointer to __shared__ = the AS(3) byte offset
// (flat-aperture mapping).
__device__ inline unsigned lds_addr_of(const void* p) {
  return (unsigned)(unsigned long long)(size_t)p;
}

#if HAVE_TDM
// 2D TDM tile load: tile_h rows x tile_w elems (bf16), row stride in elems,
// with LDS padding (pad_icode: dwords-per-interval code, pad_acode: pad dwords code)
__device__ inline void tdm_load_2d(unsigned lds_byte_addr, const void* gsrc,
                                   unsigned tile_w, unsigned tile_h,
                                   unsigned row_stride_elems,
                                   unsigned pad_icode, unsigned pad_acode) {
  unsigned long long ga = (unsigned long long)(size_t)gsrc;
  u32x4 g0;
  g0[0] = 1u;                                            // count=1, user mode
  g0[1] = lds_byte_addr;                                 // lds_addr
  g0[2] = (unsigned)ga;                                  // global_addr[31:0]
  g0[3] = (unsigned)((ga >> 32) & 0x01FFFFFFu) | (2u << 30);  // addr[56:32] | type=2
  i32x8 g1;
  g1[0] = (int)((1u << 16)                               // data_size = 2 bytes
              | (1u << 20)                               // pad_enable
              | (pad_icode << 22)                        // pad_interval
              | (pad_acode << 25));                      // pad_amount
  g1[1] = (int)((tile_w & 0xFFFFu) << 16);               // tensor_dim0[15:0]
  g1[2] = (int)(((tile_w >> 16) & 0xFFFFu) |             // tensor_dim0[31:16]
                ((tile_h & 0xFFFFu) << 16));             // tensor_dim1[15:0]
  g1[3] = (int)(((tile_h >> 16) & 0xFFFFu) |             // tensor_dim1[31:16]
                ((tile_w & 0xFFFFu) << 16));             // tile_dim0
  g1[4] = (int)(tile_h & 0xFFFFu);                       // tile_dim1 (tile_dim2 = 0)
  g1[5] = (int)row_stride_elems;                         // tensor_dim0_stride[31:0]
  g1[6] = 0;
  g1[7] = 0;
  i32x4 z4 = {0, 0, 0, 0};
#if __clang_major__ >= 23
  i32x8 z8 = {0, 0, 0, 0, 0, 0, 0, 0};
  __builtin_amdgcn_tensor_load_to_lds(g0, g1, z4, z4, z8, 0);
#else
  __builtin_amdgcn_tensor_load_to_lds(g0, g1, z4, z4, 0);
#endif
}
#endif  // HAVE_TDM

// ---------------------------------------------------------------------------
// Tiled bf16 WMMA GEMM:  Out[M=4096, N=1024] = A[4096,1024] @ Bw[1024,1024] + bias
// Block: 256 threads (8 waves), tile 128x128; wave computes 32x64 (2x4 frags).
// OUT_MODE 0: bf16 [B,H,S,Dh] head layout.  OUT_MODE 1: fp32 flat [M,N].
// bf16-A path stages the A tile with the Tensor Data Mover (hardware LDS pad).
// ---------------------------------------------------------------------------
template <typename TA, int OUT_MODE>
__global__ __launch_bounds__(256)
void gemm_wmma_bf16(const TA* __restrict__ A, const float* __restrict__ Bw,
                    const float* __restrict__ bias, void* __restrict__ Out) {
  constexpr int K = D_MODEL, N = D_MODEL;
  __shared__ __bf16 Als[128 * 40];   // [m][k] stride 40 halves
  __shared__ __bf16 Bls[128 * 40];   // transposed: [n][k] stride 40

  const int tid  = threadIdx.x;
  const int lane = tid & 31;
  const int wv   = tid >> 5;
  const int hi   = (lane >> 4) & 1;
  const int ln   = lane & 15;
  const int wm   = wv & 3;
  const int wn   = wv >> 2;
  const int n0   = blockIdx.x * 128;
  const int m0   = blockIdx.y * 128;

  v8f acc[2][4];
  for (int i = 0; i < 2; ++i)
    for (int j = 0; j < 4; ++j)
      for (int r = 0; r < 8; ++r) acc[i][j][r] = 0.0f;

  for (int kt = 0; kt < K; kt += 32) {
    // ---- Stage A tile 128x32 ----
    if constexpr (!is_f32<TA>::v) {
#if HAVE_TDM
      if (wv == 0) {
        // rows 128 x 32 bf16: 64B rows (16 DW -> icode 3), 16B pad (4 DW -> acode 3)
        tdm_load_2d(lds_addr_of(&Als[0]), A + (size_t)m0 * K + kt,
                    32u, 128u, (unsigned)K, 3u, 3u);
        __builtin_amdgcn_s_wait_tensorcnt(0);
      }
#else
      for (int c = tid; c < 512; c += 256) {
        int m = c >> 2, koff = (c & 3) << 3;
        *(bf16x8*)&Als[m * 40 + koff] = load8_cvt(A + (size_t)(m0 + m) * K + kt + koff);
      }
#endif
    } else {
      for (int c = tid; c < 512; c += 256) {
        int m = c >> 2, koff = (c & 3) << 3;
        *(bf16x8*)&Als[m * 40 + koff] = load8_cvt(A + (size_t)(m0 + m) * K + kt + koff);
      }
    }
    // ---- Stage B tile 32x128 transposed, converting fp32 -> bf16 ----
    for (int c = tid; c < 512; c += 256) {
      int kk = c >> 4, noff = (c & 15) << 3;
      const float* src = Bw + (size_t)(kt + kk) * N + n0 + noff;
      for (int i = 0; i < 8; ++i) Bls[(noff + i) * 40 + kk] = f2bf(src[i]);
    }
    // ---- Prefetch next k-tile into GL2 (global_prefetch_b8) ----
    if (kt + 32 < K) {
      __builtin_prefetch(A + (size_t)(m0 + (tid >> 1)) * K + kt + 32 + ((tid & 1) << 4), 0, 3);
      __builtin_prefetch(Bw + (size_t)(kt + 32 + (tid >> 4)) * N + n0 + ((tid & 15) << 3), 0, 3);
    }
    __syncthreads();

    v16bf af[2], bfv[4];
#pragma unroll
    for (int fm = 0; fm < 2; ++fm) {
      const __bf16* p = &Als[(wm * 32 + fm * 16 + ln) * 40];
      af[fm] = cat8(ld8(p + 8 * hi), ld8(p + 16 + 8 * hi));
    }
#pragma unroll
    for (int fn = 0; fn < 4; ++fn) {
      const __bf16* p = &Bls[(wn * 64 + fn * 16 + ln) * 40];
      bfv[fn] = cat8(ld8(p + 8 * hi), ld8(p + 16 + 8 * hi));
    }
#pragma unroll
    for (int fm = 0; fm < 2; ++fm)
#pragma unroll
      for (int fn = 0; fn < 4; ++fn)
        acc[fm][fn] = __builtin_amdgcn_wmma_f32_16x16x32_bf16(
            false, af[fm], false, bfv[fn], (short)0, acc[fm][fn], false, false);
    __syncthreads();
  }

#pragma unroll
  for (int fn = 0; fn < 4; ++fn) {
    const int n = n0 + wn * 64 + fn * 16 + ln;
    const float bv = bias[n];
#pragma unroll
    for (int fm = 0; fm < 2; ++fm) {
#pragma unroll
      for (int r = 0; r < 8; ++r) {
        float val = acc[fm][fn][r] + bv;
        int m = m0 + wm * 32 + fm * 16 + r + 8 * hi;
        if constexpr (OUT_MODE == 0) {
          int b = m >> 11, s = m & 2047, h = n >> 6, dh = n & 63;
          size_t off = ((size_t)(b * HEADS + h) * SEQ + s) * DH + dh;
          ((__bf16*)Out)[off] = f2bf(val);
        } else {
          ((float*)Out)[(size_t)m * N + n] = val;
        }
      }
    }
  }
}

// ---------------------------------------------------------------------------
// Flash attention (bf16 WMMA, fp32 online softmax).
// Block: 128 threads (4 waves) -> 64 query rows of one (batch, head).
// K/V tiles staged by the Tensor Data Mover, double-buffered: wave 0 issues
// the next tile's descriptors, then s_wait_tensorcnt(2) covers only the
// current tile so the DMA of tile i+1 overlaps WMMA on tile i.
// ---------------------------------------------------------------------------
__global__ __launch_bounds__(128)
void flash_attn_bf16(const __bf16* __restrict__ Qh, const __bf16* __restrict__ Kh,
                     const __bf16* __restrict__ Vh, __bf16* __restrict__ attn) {
  __shared__ __bf16 Kls[2][64 * 72];     // [key][d], stride 72 (TDM hw-padded)
  __shared__ __bf16 Vls[2][64 * 72];     // [key][d] raw, staged by TDM
  __shared__ __bf16 Vtls[64 * 72];       // transposed [d][key] for B-fragments
  __shared__ __bf16 Pls[4 * 16 * 72];    // per-wave P strip [row][key]

  const int tid  = threadIdx.x;
  const int lane = tid & 31;
  const int wv   = tid >> 5;
  const int hi   = (lane >> 4) & 1;
  const int ln   = lane & 15;
  const int q0   = blockIdx.x * 64;
  const int bh   = blockIdx.y;
  const int b    = bh / HEADS, h = bh % HEADS;

  const __bf16* Qp = Qh + (size_t)bh * SEQ * DH;
  const __bf16* Kp = Kh + (size_t)bh * SEQ * DH;
  const __bf16* Vp = Vh + (size_t)bh * SEQ * DH;

  v16bf qf[2];
  {
    const __bf16* qrow = Qp + (size_t)(q0 + wv * 16 + ln) * DH;
#pragma unroll
    for (int s = 0; s < 2; ++s)
      qf[s] = cat8(ld8(qrow + 32 * s + 8 * hi), ld8(qrow + 32 * s + 16 + 8 * hi));
  }

  v8f oacc[4];
  float mrow[8], lrow[8];
  for (int of = 0; of < 4; ++of)
    for (int r = 0; r < 8; ++r) oacc[of][r] = 0.0f;
  for (int r = 0; r < 8; ++r) { mrow[r] = -1e30f; lrow[r] = 0.0f; }

  __bf16* Pw = &Pls[wv * 16 * 72];
  constexpr int NT = SEQ / 64;

#if HAVE_TDM
  if (wv == 0) {
    // 64x64 bf16 rows: 128B (32 DW -> icode 4), 16B pad (4 DW -> acode 3)
    tdm_load_2d(lds_addr_of(&Kls[0][0]), Kp, 64u, 64u, (unsigned)DH, 4u, 3u);
    tdm_load_2d(lds_addr_of(&Vls[0][0]), Vp, 64u, 64u, (unsigned)DH, 4u, 3u);
  }
#endif

  for (int it = 0; it < NT; ++it) {
    const int cur = it & 1;
    const int j0  = it * 64;

#if HAVE_TDM
    if (wv == 0) {
      if (it + 1 < NT) {
        tdm_load_2d(lds_addr_of(&Kls[cur ^ 1][0]), Kp + (size_t)(j0 + 64) * DH,
                    64u, 64u, (unsigned)DH, 4u, 3u);
        tdm_load_2d(lds_addr_of(&Vls[cur ^ 1][0]), Vp + (size_t)(j0 + 64) * DH,
                    64u, 64u, (unsigned)DH, 4u, 3u);
        __builtin_amdgcn_s_wait_tensorcnt(2);   // current tile complete
      } else {
        __builtin_amdgcn_s_wait_tensorcnt(0);
      }
    }
    __syncthreads();
#else
    for (int c = tid; c < 512; c += 128) {
      int key = c >> 3, doff = (c & 7) << 3;
      *(bf16x8*)&Kls[cur][key * 72 + doff] = ld8(Kp + (size_t)(j0 + key) * DH + doff);
      *(bf16x8*)&Vls[cur][key * 72 + doff] = ld8(Vp + (size_t)(j0 + key) * DH + doff);
    }
    __syncthreads();
#endif

    // Transpose V tile in LDS: [key][d] -> [d][key]
    // (hardware alternative: ds_load_tr16_b128 directly into B-fragments)
    for (int c = tid; c < 512; c += 128) {
      int key = c >> 3, doff = (c & 7) << 3;
      bf16x8 vvv = ld8(&Vls[cur][key * 72 + doff]);
      for (int i = 0; i < 8; ++i) Vtls[(doff + i) * 72 + key] = vvv[i];
    }
    __syncthreads();

    // S = Q K^T * scale
    v8f sc[4];
#pragma unroll
    for (int c = 0; c < 4; ++c) {
      v8f a;
      for (int r = 0; r < 8; ++r) a[r] = 0.0f;
#pragma unroll
      for (int s = 0; s < 2; ++s) {
        const __bf16* p = &Kls[cur][(16 * c + ln) * 72 + 32 * s];
        v16bf kf = cat8(ld8(p + 8 * hi), ld8(p + 16 + 8 * hi));
        a = __builtin_amdgcn_wmma_f32_16x16x32_bf16(false, qf[s], false, kf,
                                                    (short)0, a, false, false);
      }
      sc[c] = a * 0.125f;
    }

    // Online softmax
    float tmax[8];
    for (int r = 0; r < 8; ++r)
      tmax[r] = fmaxf(fmaxf(sc[0][r], sc[1][r]), fmaxf(sc[2][r], sc[3][r]));
#pragma unroll
    for (int off = 1; off < 16; off <<= 1)
      for (int r = 0; r < 8; ++r)
        tmax[r] = fmaxf(tmax[r], __shfl_xor(tmax[r], off, 32));

    for (int r = 0; r < 8; ++r) {
      float mnew = fmaxf(mrow[r], tmax[r]);
      float alpha = __expf(mrow[r] - mnew);
      mrow[r] = mnew;
      lrow[r] *= alpha;
      for (int of = 0; of < 4; ++of) oacc[of][r] *= alpha;
    }

    float tsum[8];
    for (int r = 0; r < 8; ++r) tsum[r] = 0.0f;
    for (int c = 0; c < 4; ++c)
      for (int r = 0; r < 8; ++r) {
        float p = __expf(sc[c][r] - mrow[r]);
        sc[c][r] = p;
        tsum[r] += p;
      }
#pragma unroll
    for (int off = 1; off < 16; off <<= 1)
      for (int r = 0; r < 8; ++r) tsum[r] += __shfl_xor(tsum[r], off, 32);
    for (int r = 0; r < 8; ++r) lrow[r] += tsum[r];

    // C-layout P -> LDS (per-wave strip, wave-ordered DS ops)
    for (int c = 0; c < 4; ++c)
      for (int r = 0; r < 8; ++r)
        Pw[(r + 8 * hi) * 72 + 16 * c + ln] = f2bf(sc[c][r]);

    // O += P @ V
#pragma unroll
    for (int s = 0; s < 2; ++s) {
      const __bf16* pp = &Pw[ln * 72 + 32 * s];
      v16bf pf = cat8(ld8(pp + 8 * hi), ld8(pp + 16 + 8 * hi));
#pragma unroll
      for (int of = 0; of < 4; ++of) {
        const __bf16* vp = &Vtls[(16 * of + ln) * 72 + 32 * s];
        v16bf vf = cat8(ld8(vp + 8 * hi), ld8(vp + 16 + 8 * hi));
        oacc[of] = __builtin_amdgcn_wmma_f32_16x16x32_bf16(false, pf, false, vf,
                                                           (short)0, oacc[of], false, false);
      }
    }
    __syncthreads();
  }

  // Normalize and write concat directly as [B,S,D] bf16
#pragma unroll
  for (int of = 0; of < 4; ++of) {
#pragma unroll
    for (int r = 0; r < 8; ++r) {
      float val = oacc[of][r] / lrow[r];
      int srow = q0 + wv * 16 + r + 8 * hi;
      int dh = 16 * of + ln;
      size_t off = ((size_t)b * SEQ + srow) * D_MODEL + (size_t)h * DH + dh;
      attn[off] = f2bf(val);
    }
  }
}

extern "C" void kernel_launch(void* const* d_in, const int* in_sizes, int n_in,
                              void* d_out, int out_size, void* d_ws, size_t ws_size,
                              hipStream_t stream) {
  const float* q   = (const float*)d_in[0];
  const float* k   = (const float*)d_in[1];
  const float* v   = (const float*)d_in[2];
  const float* w_q = (const float*)d_in[3];
  const float* b_q = (const float*)d_in[4];
  const float* w_k = (const float*)d_in[5];
  const float* b_k = (const float*)d_in[6];
  const float* w_v = (const float*)d_in[7];
  const float* b_v = (const float*)d_in[8];
  const float* w_o = (const float*)d_in[9];
  const float* b_o = (const float*)d_in[10];
  float* out = (float*)d_out;

  // Workspace: Qh | Kh | Vh (head layout) | attn (flat), all bf16 (4 x 8 MB)
  const size_t HE = (size_t)BATCH * HEADS * SEQ * DH;
  __bf16* Qh   = (__bf16*)d_ws;
  __bf16* Kh   = Qh + HE;
  __bf16* Vh   = Kh + HE;
  __bf16* attn = Vh + HE;

  dim3 gb(256);
  dim3 gg(D_MODEL / 128, MROWS / 128);   // (8, 32)
  gemm_wmma_bf16<float, 0><<<gg, gb, 0, stream>>>(q, w_q, b_q, Qh);
  gemm_wmma_bf16<float, 0><<<gg, gb, 0, stream>>>(k, w_k, b_k, Kh);
  gemm_wmma_bf16<float, 0><<<gg, gb, 0, stream>>>(v, w_v, b_v, Vh);

  dim3 ab(128);
  dim3 ag(SEQ / 64, BATCH * HEADS);      // (32, 32)
  flash_attn_bf16<<<ag, ab, 0, stream>>>(Qh, Kh, Vh, attn);

  gemm_wmma_bf16<__bf16, 1><<<gg, gb, 0, stream>>>(attn, w_o, b_o, out);
}
